// WassVLADLayer_64269890617658
// MI455X (gfx1250) — matrix operand order repaired and checked by reading, not compiled, not run on hardware
//
#include <hip/hip_runtime.h>
#include <hip/hip_bf16.h>
#include <math.h>

typedef __attribute__((ext_vector_type(16))) __bf16 v16bf;
typedef __attribute__((ext_vector_type(8)))  float  v8f;
typedef __attribute__((ext_vector_type(16))) float  v16f;

union FragAB { v16bf v; unsigned short u[16]; uint4 q[2]; };
union FragC  { v8f v; float f[8]; };

#define B_  16
#define D_  512
#define N_  4096
#define K_  64
#define EPSF 1e-6f

// workspace byte offsets
#define WS_A1      0x0u        // 64*512 bf16 = 64 KB
#define WS_A2      0x10000u    // 64 KB
#define WS_TM2     0x20000u    // 64 f32
#define WS_LOGDET  0x20400u    // 64 f32
#define WS_PSUM    0x20800u    // 16*64 f32
#define WS_ROWNRM  0x21800u    // 16 f32
#define WS_PROBS   0x40000u    // 16*64*4096 bf16 = 8 MB
#define WS_WMU     0x840000u   // 16*512*64 f32 = 2 MB
#define WS_WX2     0x1040000u  // 2 MB

__device__ __forceinline__ unsigned short bf16_bits(float f) {
  __bf16 h = (__bf16)f;                 // native RNE cvt on gfx1250
  unsigned short b;
  __builtin_memcpy(&b, &h, 2);
  return b;
}

__global__ void gmm_init(float* psum, float* rownrm) {
  int t = blockIdx.x * blockDim.x + threadIdx.x;
  if (t < B_ * K_) psum[t] = 0.f;
  if (t < B_) rownrm[t] = 0.f;
}

__global__ __launch_bounds__(256) void gmm_prep(
    const float* __restrict__ mu, const float* __restrict__ sigma,
    unsigned short* __restrict__ A1, unsigned short* __restrict__ A2,
    float* __restrict__ tm2, float* __restrict__ logdet) {
  __shared__ float red[256];
  const int t = threadIdx.x;
  const int k = blockIdx.x;
  float m2 = 0.f, ld = 0.f;
  for (int d = t; d < D_; d += 256) {
    float sg  = fmaxf(sigma[d * K_ + k], EPSF);
    float mv  = mu[d * K_ + k];
    float inv = 1.f / (sg * sg);
    A1[k * D_ + d] = bf16_bits(inv);
    A2[k * D_ + d] = bf16_bits(mv * inv);
    m2 += mv * mv * inv;
    ld += logf(sg);
  }
  red[t] = m2; __syncthreads();
  for (int o = 128; o > 0; o >>= 1) { if (t < o) red[t] += red[t + o]; __syncthreads(); }
  if (t == 0) tm2[k] = red[0];
  __syncthreads();
  red[t] = ld; __syncthreads();
  for (int o = 128; o > 0; o >>= 1) { if (t < o) red[t] += red[t + o]; __syncthreads(); }
  if (t == 0) logdet[k] = red[0];
}

// Phase 1: logits[K=64, 32 cols] = A(K x d) * x(d x n), then softmax over K.
// Double-buffered LDS x-tile (bf16, transposed); x^2 fragment built with
// packed bf16 multiplies from the x fragment.
__global__ __launch_bounds__(256) void gmm_logits_probs(
    const float* __restrict__ x, const unsigned short* __restrict__ A1,
    const unsigned short* __restrict__ A2, const float* __restrict__ tm2,
    const float* __restrict__ logdet, const float* __restrict__ alpha,
    unsigned short* __restrict__ probs, float* __restrict__ psum) {
  __shared__ unsigned short xsT[2][32 * 40];   // [buf][n][d] bf16, padded stride
  __shared__ float lg[64 * 33];
  __shared__ float colmax[32];
  __shared__ float colinv[32];

  const int t    = threadIdx.x;
  const int lane = t & 31;
  const int w    = t >> 5;
  const int b    = blockIdx.x >> 7;           // 128 n-tiles per batch
  const int n0   = (blockIdx.x & 127) * 32;

  const int ktile = w >> 1;                   // 4 K-row tiles
  const int ntile = w & 1;                    // 2 n-col tiles
  const int lrow  = lane & 15;
  const int kh    = lane >> 4;

  FragC accX2, accXM;
  #pragma unroll
  for (int i = 0; i < 8; ++i) { accX2.f[i] = 0.f; accXM.f[i] = 0.f; }

  const int drow = t >> 3;
  const int c0   = (t & 7) * 4;
  const float* xb = x + (size_t)b * D_ * N_ + n0;

  // preload tile 0 into buffer 0
  {
    float4 xv = *reinterpret_cast<const float4*>(xb + (size_t)drow * N_ + c0);
    float vv[4] = {xv.x, xv.y, xv.z, xv.w};
    #pragma unroll
    for (int j = 0; j < 4; ++j) xsT[0][(c0 + j) * 40 + drow] = bf16_bits(vv[j]);
  }
  __syncthreads();

  for (int s = 0; s < D_ / 32; ++s) {
    const int cb = s & 1;
    // issue next tile's global load early (hides latency behind WMMA work)
    float4 nxt;
    if (s + 1 < D_ / 32)
      nxt = *reinterpret_cast<const float4*>(xb + (size_t)((s + 1) * 32 + drow) * N_ + c0);

    FragAB a1, a2, bx, bx2;
    const unsigned short* p1 = A1 + (16 * ktile + lrow) * D_ + s * 32 + 8 * kh;
    const unsigned short* p2 = A2 + (16 * ktile + lrow) * D_ + s * 32 + 8 * kh;
    a1.q[0] = *reinterpret_cast<const uint4*>(p1);
    a1.q[1] = *reinterpret_cast<const uint4*>(p1 + 16);
    a2.q[0] = *reinterpret_cast<const uint4*>(p2);
    a2.q[1] = *reinterpret_cast<const uint4*>(p2 + 16);

    const unsigned short* q1 = &xsT[cb][(16 * ntile + lrow) * 40 + 8 * kh];
    bx.q[0] = *reinterpret_cast<const uint4*>(q1);
    bx.q[1] = *reinterpret_cast<const uint4*>(q1 + 16);
    bx2.v = bx.v * bx.v;                       // v_pk_mul_bf16

    accX2.v = __builtin_amdgcn_wmma_f32_16x16x32_bf16(false, a1.v, false, bx2.v,
                                                      (short)0, accX2.v, false, false);
    accXM.v = __builtin_amdgcn_wmma_f32_16x16x32_bf16(false, a2.v, false, bx.v,
                                                      (short)0, accXM.v, false, false);

    if (s + 1 < D_ / 32) {
      float vv[4] = {nxt.x, nxt.y, nxt.z, nxt.w};
      #pragma unroll
      for (int j = 0; j < 4; ++j) xsT[cb ^ 1][(c0 + j) * 40 + drow] = bf16_bits(vv[j]);
    }
    __syncthreads();
  }

  // logits -> LDS
  const float aval = alpha[0];
  #pragma unroll
  for (int v = 0; v < 8; ++v) {
    int kg  = 16 * ktile + 8 * kh + v;
    int col = 16 * ntile + lrow;
    float dist = accX2.f[v] - 2.f * accXM.f[v] + tm2[kg];
    lg[kg * 33 + col] = -0.5f * dist - logdet[kg] + aval;
  }
  __syncthreads();

  // softmax over K=64 per column
  if (t < 32) {
    float m = -3.4e38f;
    for (int k = 0; k < 64; ++k) m = fmaxf(m, lg[k * 33 + t]);
    float ssum = 0.f;
    for (int k = 0; k < 64; ++k) ssum += __expf(lg[k * 33 + t] - m);
    colmax[t] = m;
    colinv[t] = 1.f / ssum;
  }
  __syncthreads();

  #pragma unroll
  for (int j = 0; j < 8; ++j) {
    int k   = (t >> 5) + 8 * j;
    int col = t & 31;
    float p = __expf(lg[k * 33 + col] - colmax[col]) * colinv[col];
    lg[k * 33 + col] = p;
    probs[(size_t)(b * 64 + k) * N_ + n0 + col] = bf16_bits(p);
  }
  __syncthreads();
  if (t < 64) {
    float s = 0.f;
    for (int c = 0; c < 32; ++c) s += lg[t * 33 + c];
    atomicAdd(&psum[b * 64 + t], s);
  }
}

// Phase 2: weighted[d,k] = sum_n x[d,n]*probs[k,n]; per-k scale folded in at the end.
__global__ __launch_bounds__(256) void gmm_aggregate(
    const float* __restrict__ x, const unsigned short* __restrict__ probs,
    const float* __restrict__ psum, float* __restrict__ wmu, float* __restrict__ wx2) {
  const int t    = threadIdx.x;
  const int lane = t & 31;
  const int w    = t >> 5;
  const int W    = blockIdx.x * 8 + w;   // 2048 wave-tiles = 16b * 32dtile * 4ktile
  const int b     = W >> 7;
  const int r     = W & 127;
  const int dtile = r >> 2;
  const int ktile = r & 3;
  const int lrow  = lane & 15;
  const int kh    = lane >> 4;

  FragC accMu, accX2;
  #pragma unroll
  for (int i = 0; i < 8; ++i) { accMu.f[i] = 0.f; accX2.f[i] = 0.f; }

  const float* xr = x + ((size_t)b * D_ + 16 * dtile + lrow) * N_;
  const unsigned short* pr = probs + (size_t)(b * 64 + 16 * ktile + lrow) * N_ + 8 * kh;

  for (int s = 0; s < N_ / 32; ++s) {
    const int n0 = s * 32;
    FragAB xa, x2a, pf;
    float4 c0 = *reinterpret_cast<const float4*>(xr + n0 + 8 * kh);
    float4 c1 = *reinterpret_cast<const float4*>(xr + n0 + 8 * kh + 4);
    float4 c2 = *reinterpret_cast<const float4*>(xr + n0 + 16 + 8 * kh);
    float4 c3 = *reinterpret_cast<const float4*>(xr + n0 + 16 + 8 * kh + 4);
    v16f fv;
    fv[0] = c0.x;  fv[1] = c0.y;  fv[2]  = c0.z;  fv[3]  = c0.w;
    fv[4] = c1.x;  fv[5] = c1.y;  fv[6]  = c1.z;  fv[7]  = c1.w;
    fv[8] = c2.x;  fv[9] = c2.y;  fv[10] = c2.z;  fv[11] = c2.w;
    fv[12] = c3.x; fv[13] = c3.y; fv[14] = c3.z;  fv[15] = c3.w;
    xa.v  = __builtin_convertvector(fv, v16bf);  // v_cvt_pk_bf16_f32
    x2a.v = xa.v * xa.v;                         // v_pk_mul_bf16

    pf.q[0] = *reinterpret_cast<const uint4*>(pr + n0);
    pf.q[1] = *reinterpret_cast<const uint4*>(pr + n0 + 16);

    accMu.v = __builtin_amdgcn_wmma_f32_16x16x32_bf16(false, xa.v,  false, pf.v,
                                                      (short)0, accMu.v, false, false);
    accX2.v = __builtin_amdgcn_wmma_f32_16x16x32_bf16(false, x2a.v, false, pf.v,
                                                      (short)0, accX2.v, false, false);
  }

  const int kcol  = 16 * ktile + lrow;
  const float inv = 1.f / (psum[b * 64 + kcol] + EPSF);
  #pragma unroll
  for (int v = 0; v < 8; ++v) {
    int dg = 16 * dtile + 8 * kh + v;
    size_t o = ((size_t)b * D_ + dg) * K_ + kcol;
    wmu[o] = accMu.f[v] * inv;
    wx2[o] = accX2.f[v] * inv;
  }
}

__global__ __launch_bounds__(256) void gmm_finalize1(
    const float* __restrict__ wmu, const float* __restrict__ wx2,
    const float* __restrict__ mu, const float* __restrict__ sigma,
    float* __restrict__ out, float* __restrict__ rownrm) {
  __shared__ float red[256];
  const int t = threadIdx.x;
  const int b = blockIdx.x >> 6;
  const int k = blockIdx.x & 63;

  float dmu[2], dsg[2], ss = 0.f;
  #pragma unroll
  for (int j = 0; j < 2; ++j) {
    int d = t + j * 256;
    size_t o = ((size_t)b * D_ + d) * K_ + k;
    float wm = wmu[o], w2 = wx2[o];
    float m  = mu[d * K_ + k], sg = sigma[d * K_ + k];
    float sh = sqrtf(fmaxf(w2 - wm * wm, EPSF));
    dmu[j] = wm - m;
    dsg[j] = sh - sg;
    ss += dmu[j] * dmu[j] + dsg[j] * dsg[j];
  }
  red[t] = ss; __syncthreads();
  for (int o = 128; o > 0; o >>= 1) { if (t < o) red[t] += red[t + o]; __syncthreads(); }
  float total = red[0];
  float inv = 1.f / fmaxf(sqrtf(total), EPSF);
  #pragma unroll
  for (int j = 0; j < 2; ++j) {
    int d = t + j * 256;
    out[(size_t)b * (2 * D_ * K_) + (size_t)d * K_ + k]         = dmu[j] * inv;
    out[(size_t)b * (2 * D_ * K_) + (size_t)(D_ + d) * K_ + k]  = dsg[j] * inv;
  }
  if (t == 0) atomicAdd(&rownrm[b], total * inv * inv);
}

__global__ void gmm_finalize2(float* out, const float* __restrict__ rownrm) {
  size_t i = (size_t)blockIdx.x * blockDim.x + threadIdx.x;
  int b = (int)(i >> 16);   // 2*D*K = 65536 per batch
  float inv = 1.f / fmaxf(sqrtf(rownrm[b]), EPSF);
  out[i] *= inv;
}

extern "C" void kernel_launch(void* const* d_in, const int* in_sizes, int n_in,
                              void* d_out, int out_size, void* d_ws, size_t ws_size,
                              hipStream_t stream) {
  const float* x     = (const float*)d_in[0];
  const float* mu    = (const float*)d_in[1];
  const float* sigma = (const float*)d_in[2];
  const float* alpha = (const float*)d_in[3];
  float* out = (float*)d_out;
  char*  ws  = (char*)d_ws;

  unsigned short* A1 = (unsigned short*)(ws + WS_A1);
  unsigned short* A2 = (unsigned short*)(ws + WS_A2);
  float* tm2    = (float*)(ws + WS_TM2);
  float* logdet = (float*)(ws + WS_LOGDET);
  float* psum   = (float*)(ws + WS_PSUM);
  float* rownrm = (float*)(ws + WS_ROWNRM);
  unsigned short* probs = (unsigned short*)(ws + WS_PROBS);
  float* wmu = (float*)(ws + WS_WMU);
  float* wx2 = (float*)(ws + WS_WX2);

  gmm_init<<<4, 256, 0, stream>>>(psum, rownrm);
  gmm_prep<<<K_, 256, 0, stream>>>(mu, sigma, A1, A2, tm2, logdet);
  gmm_logits_probs<<<B_ * (N_ / 32), 256, 0, stream>>>(x, A1, A2, tm2, logdet, alpha,
                                                       probs, psum);
  gmm_aggregate<<<256, 256, 0, stream>>>(x, probs, psum, wmu, wx2);
  gmm_finalize1<<<B_ * K_, 256, 0, stream>>>(wmu, wx2, mu, sigma, out, rownrm);
  gmm_finalize2<<<4096, 256, 0, stream>>>(out, rownrm);
}
